// Model_pool_norm_13511967113356
// MI455X (gfx1250) — matrix operand order repaired and checked by reading, not compiled, not run on hardware
//
#include <hip/hip_runtime.h>
#include <hip/hip_bf16.h>

// ---------------------------------------------------------------------------
// Model sizes (fixed by the reference)
// ---------------------------------------------------------------------------
#define BGR   64
#define NPG   330
#define EPG   5280
#define IN_C  64
#define HID   256
#define K1    165
#define K2    83
#define NTOT  (BGR*NPG)     // 21120
#define ETOT  (BGR*EPG)     // 337920

typedef __attribute__((ext_vector_type(16))) __bf16 v16bf;
typedef __attribute__((ext_vector_type(8)))  float  v8f;

// ---------------------------------------------------------------------------
// Tile loaders for WMMA bf16 (CDNA5 ISA 7.12.2 layouts, wave32).
// `fast` is wave-uniform: interior tiles take unguarded contiguous loads.
// Edge tiles: clamped addresses + multiply-by-mask so the loaded value is
// ALWAYS used -> clang cannot sink loads under exec-mask branches.
// ---------------------------------------------------------------------------
template<int TRANSA>
__device__ __forceinline__ v16bf load_a_tile(const float* __restrict__ A, int lda,
                                             int mrow, int k0, int half,
                                             int M, int K, bool fast)
{
    v16bf a;
    if (fast) {
        if (TRANSA) {
#pragma unroll
            for (int i = 0; i < 16; ++i) {
                int kk = k0 + ((i >> 3) << 4) + (half << 3) + (i & 7);
                a[i] = (__bf16)A[(long)kk * lda + mrow];
            }
        } else {
            const float* p = A + (long)mrow * lda + k0 + (half << 3);
#pragma unroll
            for (int i = 0; i < 8; ++i) a[i]     = (__bf16)p[i];       // K = half*8 + i
#pragma unroll
            for (int i = 0; i < 8; ++i) a[8 + i] = (__bf16)p[16 + i];  // K = 16 + half*8 + i
        }
    } else {
        const int   mc   = mrow < M ? mrow : 0;
        const float mmask = mrow < M ? 1.f : 0.f;
#pragma unroll
        for (int i = 0; i < 16; ++i) {
            int kk = k0 + ((i >> 3) << 4) + (half << 3) + (i & 7);
            int kc = kk < K ? kk : 0;
            float sel = kk < K ? mmask : 0.f;
            float v = TRANSA ? A[(long)kc * lda + mc] : A[(long)mc * lda + kc];
            a[i] = (__bf16)(v * sel);        // load always consumed -> no branches
        }
    }
    return a;
}

__device__ __forceinline__ v16bf load_b_tile(const float* __restrict__ Bm, int ldb,
                                             int ncol, int k0, int half,
                                             int Nc, int K, bool fast)
{
    v16bf b;
    if (fast) {
        const float* p = Bm + (long)(k0 + (half << 3)) * ldb + ncol;
#pragma unroll
        for (int i = 0; i < 16; ++i) b[i] = (__bf16)p[(long)(i + ((i >> 3) << 3)) * ldb];  // K = 16*(i/8) + half*8 + i%8
    } else {
        const int   nc2   = ncol < Nc ? ncol : 0;
        const float nmask = ncol < Nc ? 1.f : 0.f;
#pragma unroll
        for (int i = 0; i < 16; ++i) {
            int kk = k0 + ((i >> 3) << 4) + (half << 3) + (i & 7);
            int kc = kk < K ? kk : 0;
            float sel = kk < K ? nmask : 0.f;
            b[i] = (__bf16)(Bm[(long)kc * ldb + nc2] * sel);
        }
    }
    return b;
}

__device__ __forceinline__ v8f load_c_tile(const float* __restrict__ C, int ldc,
                                           int mt, int nt, int M, int Nc,
                                           int half, int l16)
{
    v8f c = {};
    if ((mt + 16) <= M && (nt + 16) <= Nc) {
        const float* p = C + (long)(mt + half * 8) * ldc + nt + l16;
#pragma unroll
        for (int r = 0; r < 8; ++r) c[r] = p[(long)r * ldc];
    } else {
        int ni = nt + l16;
        int nc2 = ni < Nc ? ni : 0;
        float nmask = ni < Nc ? 1.f : 0.f;
#pragma unroll
        for (int r = 0; r < 8; ++r) {
            int mi = mt + half * 8 + r;
            int mc = mi < M ? mi : 0;
            float sel = mi < M ? nmask : 0.f;
            c[r] = C[(long)mc * ldc + nc2] * sel;
        }
    }
    return c;
}

__device__ __forceinline__ void store_c_tile(float* __restrict__ C, int ldc,
                                             const v8f& c, int mt, int nt,
                                             int M, int Nc, const float* bias,
                                             int flags, int half, int l16)
{
    int ni = nt + l16;
    float badd = (bias && ni < Nc) ? bias[ni] : 0.f;
    if ((mt + 16) <= M && (nt + 16) <= Nc) {
        float* p = C + (long)(mt + half * 8) * ldc + ni;
#pragma unroll
        for (int r = 0; r < 8; ++r) {
            float o = c[r] + badd;
            if (flags & 2) o = fmaxf(o, 0.f);
            p[(long)r * ldc] = o;
        }
    } else {
#pragma unroll
        for (int r = 0; r < 8; ++r) {
            int mi = mt + half * 8 + r;
            if (mi < M && ni < Nc) {
                float o = c[r] + badd;
                if (flags & 2) o = fmaxf(o, 0.f);
                C[(long)mi * ldc + ni] = o;
            }
        }
    }
}

// ---------------------------------------------------------------------------
// Batched GEMM on WMMA bf16 tensor cores, f32 accumulate.
//   C[b] = act( (flags&1 ? C[b] : 0) + opA(A[b]) @ B[b] + bias )
// One wave computes a 32x32 C block (2x2 WMMA tiles): 4 wmma per K-step,
// A/B register tiles each reused twice.
// flags: bit0 = accumulate into existing C, bit1 = ReLU at store.
// ---------------------------------------------------------------------------
template<int TRANSA>
__global__ void __launch_bounds__(32)
wmma_bgemm_t(const float* __restrict__ A, const float* __restrict__ Bm,
             const float* __restrict__ bias, float* __restrict__ C,
             int M, int Nc, int K, int lda, int ldb, int ldc,
             long sA, long sB, long sC, int flags)
{
    const int lane = threadIdx.x & 31;   // blockDim.x == 32, one wave per block
    const int half = lane >> 4;
    const int l16  = lane & 15;
    const int n0 = blockIdx.x * 32;
    const int m0 = blockIdx.y * 32;
    A  += (long)blockIdx.z * sA;
    Bm += (long)blockIdx.z * sB;
    C  += (long)blockIdx.z * sC;

    const bool mf0 = (m0 + 16) <= M;    // wave-uniform tile-validity flags
    const bool mf1 = (m0 + 32) <= M;
    const bool nf0 = (n0 + 16) <= Nc;
    const bool nf1 = (n0 + 32) <= Nc;

    v8f c00 = {}, c01 = {}, c10 = {}, c11 = {};
    if (flags & 1) {
        c00 = load_c_tile(C, ldc, m0,      n0,      M, Nc, half, l16);
        c01 = load_c_tile(C, ldc, m0,      n0 + 16, M, Nc, half, l16);
        c10 = load_c_tile(C, ldc, m0 + 16, n0,      M, Nc, half, l16);
        c11 = load_c_tile(C, ldc, m0 + 16, n0 + 16, M, Nc, half, l16);
    }

    for (int k0 = 0; k0 < K; k0 += 32) {
        const bool kf = (k0 + 32) <= K;
        v16bf a0 = load_a_tile<TRANSA>(A, lda, m0 + l16,      k0, half, M, K, kf && mf0);
        v16bf a1 = load_a_tile<TRANSA>(A, lda, m0 + 16 + l16, k0, half, M, K, kf && mf1);
        v16bf b0 = load_b_tile(Bm, ldb, n0 + l16,      k0, half, Nc, K, kf && nf0);
        v16bf b1 = load_b_tile(Bm, ldb, n0 + 16 + l16, k0, half, Nc, K, kf && nf1);
        c00 = __builtin_amdgcn_wmma_f32_16x16x32_bf16(false, a0, false, b0, (short)0, c00, false, false);
        c01 = __builtin_amdgcn_wmma_f32_16x16x32_bf16(false, a0, false, b1, (short)0, c01, false, false);
        c10 = __builtin_amdgcn_wmma_f32_16x16x32_bf16(false, a1, false, b0, (short)0, c10, false, false);
        c11 = __builtin_amdgcn_wmma_f32_16x16x32_bf16(false, a1, false, b1, (short)0, c11, false, false);
    }

    store_c_tile(C, ldc, c00, m0,      n0,      M, Nc, bias, flags, half, l16);
    store_c_tile(C, ldc, c01, m0,      n0 + 16, M, Nc, bias, flags, half, l16);
    store_c_tile(C, ldc, c10, m0 + 16, n0,      M, Nc, bias, flags, half, l16);
    store_c_tile(C, ldc, c11, m0 + 16, n0 + 16, M, Nc, bias, flags, half, l16);
}

// ---------------------------------------------------------------------------
// Helpers
// ---------------------------------------------------------------------------
__device__ inline void atomicMaxFloat(float* addr, float val) {
    // works for mixed-sign values given init = -inf (0xFF800000)
    if (val >= 0.f) atomicMax((int*)addr, __float_as_int(val));
    else            atomicMin((unsigned int*)addr, __float_as_uint(val));
}

__global__ void fillk(float* p, float v, long n) {
    long i = (long)blockIdx.x * blockDim.x + threadIdx.x;
    if (i < n) p[i] = v;
}

// ---------------------------------------------------------------------------
// Stage We (5 x 256 f32 = 5120 B) into LDS with the CDNA5 async global->LDS
// pipeline: per-lane b128 copies tracked by ASYNCcnt, drained with
// s_wait_asynccnt, then published to the workgroup by the barrier.
// LDS VDST operand = low 32 bits of the flat shared-aperture address.
// ---------------------------------------------------------------------------
__device__ __forceinline__ void stage_We_lds(float* sWe, const float* __restrict__ We)
{
    for (int i = threadIdx.x; i < (5 * HID) / 4; i += blockDim.x) {   // 320 x b128
        unsigned     lds = (unsigned)(unsigned long long)(&sWe[i * 4]);
        const float* g   = We + i * 4;
        asm volatile("global_load_async_to_lds_b128 %0, %1, off"
                     :: "v"(lds), "v"(g) : "memory");
    }
    asm volatile("s_wait_asynccnt 0x0" ::: "memory");   // this wave's copies done
    __syncthreads();                                    // publish LDS to all waves
}

// ---------------------------------------------------------------------------
// Edge phase.  e = edge_attr @ We is fused (never materialized: saves ~1 GB);
// We itself is served from LDS.  One wave per edge; lanes stride the channels.
// ---------------------------------------------------------------------------
__global__ void edge_alpha_max(const int* __restrict__ ei, const float* __restrict__ ea,
                               const float* __restrict__ We,
                               const float* __restrict__ q, const float* __restrict__ k,
                               float* __restrict__ alpha, float* __restrict__ m)
{
    __shared__ float sWe[5 * HID];
    stage_We_lds(sWe, We);

    int wid  = (blockIdx.x * blockDim.x + threadIdx.x) >> 5;
    int lane = threadIdx.x & 31;
    if (wid >= ETOT) return;
    int src = ei[wid];
    int dst = ei[ETOT + wid];
    float e0 = ea[wid*5+0], e1 = ea[wid*5+1], e2 = ea[wid*5+2],
          e3 = ea[wid*5+3], e4 = ea[wid*5+4];
    float acc = 0.f;
    for (int c = lane; c < HID; c += 32) {
        float e = e0*sWe[c] + e1*sWe[HID+c] + e2*sWe[2*HID+c]
                + e3*sWe[3*HID+c] + e4*sWe[4*HID+c];
        acc += q[(long)dst*HID + c] * (k[(long)src*HID + c] + e);
    }
    for (int off = 16; off > 0; off >>= 1) acc += __shfl_down(acc, off, 32);
    if (lane == 0) {
        float al = acc * 0.0625f;                 // 1/sqrt(256)
        alpha[wid] = al;
        atomicMaxFloat(&m[dst], al);
    }
}

__global__ void edge_exp(const int* __restrict__ ei, float* __restrict__ alpha,
                         const float* __restrict__ m, float* __restrict__ denom)
{
    int e = blockIdx.x * blockDim.x + threadIdx.x;
    if (e >= ETOT) return;
    int dst = ei[ETOT + e];
    float a = __expf(alpha[e] - m[dst]);
    alpha[e] = a;
    atomicAdd(&denom[dst], a);
}

__global__ void edge_msg(const int* __restrict__ ei, const float* __restrict__ ea,
                         const float* __restrict__ We, const float* __restrict__ v,
                         const float* __restrict__ alpha, const float* __restrict__ denom,
                         float* __restrict__ h)
{
    __shared__ float sWe[5 * HID];
    stage_We_lds(sWe, We);

    int wid  = (blockIdx.x * blockDim.x + threadIdx.x) >> 5;
    int lane = threadIdx.x & 31;
    if (wid >= ETOT) return;
    int src = ei[wid];
    int dst = ei[ETOT + wid];
    float an = alpha[wid] / (denom[dst] + 1e-16f);
    float e0 = ea[wid*5+0], e1 = ea[wid*5+1], e2 = ea[wid*5+2],
          e3 = ea[wid*5+3], e4 = ea[wid*5+4];
    for (int c = lane; c < HID; c += 32) {
        float e = e0*sWe[c] + e1*sWe[HID+c] + e2*sWe[2*HID+c]
                + e3*sWe[3*HID+c] + e4*sWe[4*HID+c];
        atomicAdd(&h[(long)dst*HID + c], (v[(long)src*HID + c] + e) * an);
    }
}

// ---------------------------------------------------------------------------
// BatchNorm1d (training stats over N rows) + ReLU.  One block per channel.
// ---------------------------------------------------------------------------
__global__ void bn_relu(float* __restrict__ h, const float* __restrict__ g,
                        const float* __restrict__ bb, int n)
{
    int c = blockIdx.x;
    __shared__ float red[256];
    float s = 0.f, s2 = 0.f;
    for (int i = threadIdx.x; i < n; i += blockDim.x) {
        float v = h[(long)i*HID + c];
        s += v; s2 += v*v;
    }
    red[threadIdx.x] = s; __syncthreads();
    for (int st = 128; st > 0; st >>= 1) {
        if (threadIdx.x < st) red[threadIdx.x] += red[threadIdx.x + st];
        __syncthreads();
    }
    float mu = red[0] / n; __syncthreads();
    red[threadIdx.x] = s2; __syncthreads();
    for (int st = 128; st > 0; st >>= 1) {
        if (threadIdx.x < st) red[threadIdx.x] += red[threadIdx.x + st];
        __syncthreads();
    }
    float var = red[0] / n - mu * mu;
    float inv = rsqrtf(var + 1e-5f);
    float gg = g[c], bc = bb[c];
    for (int i = threadIdx.x; i < n; i += blockDim.x) {
        float v = h[(long)i*HID + c];
        h[(long)i*HID + c] = fmaxf((v - mu) * inv * gg + bc, 0.f);
    }
}

// ---------------------------------------------------------------------------
// Dense adjacency build (graphs contiguous, 330 nodes each)
// ---------------------------------------------------------------------------
__global__ void build_adj(const int* __restrict__ ei, float* __restrict__ adj) {
    int e = blockIdx.x * blockDim.x + threadIdx.x;
    if (e >= ETOT) return;
    int src = ei[e], dst = ei[ETOT + e];
    int b = src / NPG;
    atomicAdd(&adj[((long)b*NPG + (src - b*NPG))*NPG + (dst - b*NPG)], 1.0f);
}

// ---------------------------------------------------------------------------
// Row softmax (cluster assignments) — one thread per row
// ---------------------------------------------------------------------------
__global__ void softmax_rows(float* __restrict__ s, int rows, int k) {
    int r = blockIdx.x * blockDim.x + threadIdx.x;
    if (r >= rows) return;
    float* p = s + (long)r * k;
    float mx = -1e30f;
    for (int i = 0; i < k; ++i) mx = fmaxf(mx, p[i]);
    float sum = 0.f;
    for (int i = 0; i < k; ++i) { float e = __expf(p[i]-mx); p[i] = e; sum += e; }
    float inv = 1.f / sum;
    for (int i = 0; i < k; ++i) p[i] *= inv;
}

// ---------------------------------------------------------------------------
// mincut pooled-adjacency: zero diagonal, d = sqrt(rowsum)+eps, A /= d_i d_j
// ---------------------------------------------------------------------------
__global__ void pool_prep(float* __restrict__ adj, float* __restrict__ deg, int k) {
    int idx = blockIdx.x * blockDim.x + threadIdx.x;
    if (idx >= BGR * k) return;
    int b = idx / k, i = idx - b * k;
    float* row = adj + ((long)b*k + i) * k;
    row[i] = 0.f;
    float s = 0.f;
    for (int j = 0; j < k; ++j) s += row[j];
    deg[idx] = sqrtf(s) + 1e-15f;
}
__global__ void pool_norm(float* __restrict__ adj, const float* __restrict__ deg, int k) {
    long t = (long)blockIdx.x * blockDim.x + threadIdx.x;
    long tot = (long)BGR * k * k;
    if (t >= tot) return;
    int b = (int)(t / ((long)k*k));
    int rem = (int)(t - (long)b*k*k);
    int i = rem / k, j = rem - i*k;
    adj[t] /= (deg[b*k + i] * deg[b*k + j]);
}

// ---------------------------------------------------------------------------
// LayerNorm over HID + ReLU — one thread per row
// ---------------------------------------------------------------------------
__global__ void ln_relu(float* __restrict__ x, const float* __restrict__ g,
                        const float* __restrict__ b, int rows)
{
    int r = blockIdx.x * blockDim.x + threadIdx.x;
    if (r >= rows) return;
    float* p = x + (long)r * HID;
    float mu = 0.f;
    for (int i = 0; i < HID; ++i) mu += p[i];
    mu *= (1.f / HID);
    float var = 0.f;
    for (int i = 0; i < HID; ++i) { float d = p[i] - mu; var += d*d; }
    var *= (1.f / HID);
    float inv = rsqrtf(var + 1e-5f);
    for (int i = 0; i < HID; ++i)
        p[i] = fmaxf((p[i] - mu) * inv * g[i] + b[i], 0.f);
}

// ---------------------------------------------------------------------------
// Readout: mean over K2 clusters of x@Wro + bro, sigmoid
// ---------------------------------------------------------------------------
__global__ void readout(const float* __restrict__ x, const float* __restrict__ Wro,
                        const float* __restrict__ bro, float* __restrict__ out)
{
    int b = blockIdx.x * blockDim.x + threadIdx.x;
    if (b >= BGR) return;
    const float* p = x + (long)b * K2 * HID;
    float acc = 0.f;
    for (int n = 0; n < K2; ++n)
        for (int c = 0; c < HID; ++c)
            acc += p[n*HID + c] * Wro[c];
    float logit = acc / (float)K2 + bro[0];
    out[b] = 1.f / (1.f + __expf(-logit));
}

// ---------------------------------------------------------------------------
// Host-side launch sequence
// ---------------------------------------------------------------------------
static inline void gemm(hipStream_t st, const float* A, const float* Bm, const float* bias,
                        float* C, int M, int Nc, int K, int lda, int ldb, int ldc,
                        long sA, long sB, long sC, int batch, int transA, int flags)
{
    dim3 g((Nc + 31) / 32, (M + 31) / 32, batch);
    if (transA)
        wmma_bgemm_t<1><<<g, 32, 0, st>>>(A, Bm, bias, C, M, Nc, K, lda, ldb, ldc,
                                          sA, sB, sC, flags);
    else
        wmma_bgemm_t<0><<<g, 32, 0, st>>>(A, Bm, bias, C, M, Nc, K, lda, ldb, ldc,
                                          sA, sB, sC, flags);
}

extern "C" void kernel_launch(void* const* d_in, const int* in_sizes, int n_in,
                              void* d_out, int out_size, void* d_ws, size_t ws_size,
                              hipStream_t stream)
{
    (void)in_sizes; (void)n_in; (void)out_size; (void)ws_size;
    const float* x    = (const float*)d_in[0];
    const int*   ei   = (const int*)  d_in[1];
    const float* ea   = (const float*)d_in[2];
    // d_in[3] = batch (unused; graphs contiguous)
    const float* Wq   = (const float*)d_in[4];  const float* bq   = (const float*)d_in[5];
    const float* Wk   = (const float*)d_in[6];  const float* bk   = (const float*)d_in[7];
    const float* Wv   = (const float*)d_in[8];  const float* bv   = (const float*)d_in[9];
    const float* We   = (const float*)d_in[10];
    const float* Wskip= (const float*)d_in[11]; const float* bskip= (const float*)d_in[12];
    const float* bng  = (const float*)d_in[13]; const float* bnb  = (const float*)d_in[14];
    const float* Wm1  = (const float*)d_in[15]; const float* bm1  = (const float*)d_in[16];
    const float* Wrel2= (const float*)d_in[17]; const float* brel2= (const float*)d_in[18];
    const float* Wroot2=(const float*)d_in[19];
    const float* ln2g = (const float*)d_in[20]; const float* ln2b = (const float*)d_in[21];
    const float* Wm2  = (const float*)d_in[22]; const float* bm2  = (const float*)d_in[23];
    const float* Wrel3= (const float*)d_in[24]; const float* brel3= (const float*)d_in[25];
    const float* Wroot3=(const float*)d_in[26];
    const float* ln3g = (const float*)d_in[27]; const float* ln3b = (const float*)d_in[28];
    const float* Wl1  = (const float*)d_in[29]; const float* bl1  = (const float*)d_in[30];
    const float* Wro  = (const float*)d_in[31]; const float* bro  = (const float*)d_in[32];
    float* out = (float*)d_out;
    float* ws  = (float*)d_ws;

    // ---- workspace layout (floats), phase-based reuse; peak ~126 MB ----
    const long NQ = (long)NTOT * HID;                 // 5,406,720
    float* q     = ws;                                // phase A
    float* kbuf  = ws + NQ;
    float* vbuf  = ws + 2*NQ;
    float* h     = ws + 3*NQ;                         // persists as dense x
    float* alpha = ws + 4*NQ;                         // ETOT
    float* mbuf  = alpha + ETOT;                      // NTOT
    float* denom = mbuf + NTOT;                       // NTOT
    // phase B reuses q/k/v region [0, 3*NQ)
    float* adj   = ws;                                // 64*330*330 = 6,969,600
    float* s1    = adj + (long)BGR*NPG*NPG;           // 64*330*165 = 3,484,800
    float* t1    = s1  + (long)BGR*NPG*K1;            // 3,484,800 (ends < 3*NQ)
    // region above h (alpha/m/denom freed after edge phase)
    float* x1    = ws + 4*NQ;                         // 64*165*256 = 2,703,360
    float* adj1  = x1   + (long)BGR*K1*HID;           // 64*165*165 = 1,742,400
    float* deg   = adj1 + (long)BGR*K1*K1;            // 64*330
    float* tmpA  = deg  + (long)BGR*NPG;              // 2,703,360
    float* tmpB  = tmpA + (long)BGR*K1*HID;           // 2,703,360
    // phase C reuses [0, adj+s1+t1) region
    float* s2    = ws;                                // 64*165*83
    float* t2    = s2   + (long)BGR*K1*K2;
    float* x2    = t2   + (long)BGR*K1*K2;            // 64*83*256
    float* adj2  = x2   + (long)BGR*K2*HID;           // 64*83*83
    float* tmpC  = adj2 + (long)BGR*K2*K2;
    float* tmpD  = tmpC + (long)BGR*K2*HID;
    float* l1o   = tmpD + (long)BGR*K2*HID;

    const int TB = 256;
    // ---- init ----
    fillk<<<(NTOT+TB-1)/TB, TB, 0, stream>>>(mbuf, -INFINITY, NTOT);
    fillk<<<(NTOT+TB-1)/TB, TB, 0, stream>>>(denom, 0.f, NTOT);

    // ---- TransformerConv projections (WMMA) ----
    gemm(stream, x, Wq,    bq,    q,    NTOT, HID, IN_C, IN_C, HID, HID, 0,0,0, 1, 0, 0);
    gemm(stream, x, Wk,    bk,    kbuf, NTOT, HID, IN_C, IN_C, HID, HID, 0,0,0, 1, 0, 0);
    gemm(stream, x, Wv,    bv,    vbuf, NTOT, HID, IN_C, IN_C, HID, HID, 0,0,0, 1, 0, 0);
    gemm(stream, x, Wskip, bskip, h,    NTOT, HID, IN_C, IN_C, HID, HID, 0,0,0, 1, 0, 0);

    // ---- edge attention (fused edge-MLP, LDS-staged We), softmax, scatter ----
    {
        long thr = (long)ETOT * 32;
        int blocks = (int)((thr + TB - 1) / TB);
        edge_alpha_max<<<blocks, TB, 0, stream>>>(ei, ea, We, q, kbuf, alpha, mbuf);
        edge_exp<<<(ETOT+TB-1)/TB, TB, 0, stream>>>(ei, alpha, mbuf, denom);
        edge_msg<<<blocks, TB, 0, stream>>>(ei, ea, We, vbuf, alpha, denom, h);
    }

    // ---- BatchNorm + ReLU ----
    bn_relu<<<HID, TB, 0, stream>>>(h, bng, bnb, NTOT);

    // ---- dense adjacency ----
    fillk<<<(int)(((long)BGR*NPG*NPG+TB-1)/TB), TB, 0, stream>>>(adj, 0.f, (long)BGR*NPG*NPG);
    build_adj<<<(ETOT+TB-1)/TB, TB, 0, stream>>>(ei, adj);

    // ---- mincut pool 1 ----
    gemm(stream, h, Wm1, bm1, s1, NTOT, K1, HID, HID, K1, K1, 0,0,0, 1, 0, 0);
    softmax_rows<<<(NTOT+TB-1)/TB, TB, 0, stream>>>(s1, NTOT, K1);
    gemm(stream, s1, h, nullptr, x1, K1, HID, NPG, K1, HID, HID,
         (long)NPG*K1, (long)NPG*HID, (long)K1*HID, BGR, 1, 0);          // x1 = s^T x
    gemm(stream, adj, s1, nullptr, t1, NPG, K1, NPG, NPG, K1, K1,
         (long)NPG*NPG, (long)NPG*K1, (long)NPG*K1, BGR, 0, 0);          // t1 = A s
    gemm(stream, s1, t1, nullptr, adj1, K1, K1, NPG, K1, K1, K1,
         (long)NPG*K1, (long)NPG*K1, (long)K1*K1, BGR, 1, 0);            // adj1 = s^T A s
    pool_prep<<<(BGR*K1+TB-1)/TB, TB, 0, stream>>>(adj1, deg, K1);
    pool_norm<<<(int)(((long)BGR*K1*K1+TB-1)/TB), TB, 0, stream>>>(adj1, deg, K1);

    // ---- DenseGraphConv 2 + LayerNorm + ReLU ----
    gemm(stream, adj1, x1, nullptr, tmpA, K1, HID, K1, K1, HID, HID,
         (long)K1*K1, (long)K1*HID, (long)K1*HID, BGR, 0, 0);            // A x
    gemm(stream, x1, Wroot2, nullptr, tmpB, K1, HID, HID, HID, HID, HID,
         (long)K1*HID, 0, (long)K1*HID, BGR, 0, 0);                      // x Wroot
    gemm(stream, tmpA, Wrel2, brel2, tmpB, K1, HID, HID, HID, HID, HID,
         (long)K1*HID, 0, (long)K1*HID, BGR, 0, 1);                      // += (Ax)Wrel + b
    ln_relu<<<(BGR*K1+TB-1)/TB, TB, 0, stream>>>(tmpB, ln2g, ln2b, BGR*K1);

    // ---- mincut pool 2 ----
    gemm(stream, tmpB, Wm2, bm2, s2, BGR*K1, K2, HID, HID, K2, K2, 0,0,0, 1, 0, 0);
    softmax_rows<<<(BGR*K1+TB-1)/TB, TB, 0, stream>>>(s2, BGR*K1, K2);
    gemm(stream, s2, tmpB, nullptr, x2, K2, HID, K1, K2, HID, HID,
         (long)K1*K2, (long)K1*HID, (long)K2*HID, BGR, 1, 0);            // x2 = s^T x
    gemm(stream, adj1, s2, nullptr, t2, K1, K2, K1, K1, K2, K2,
         (long)K1*K1, (long)K1*K2, (long)K1*K2, BGR, 0, 0);              // t2 = A s
    gemm(stream, s2, t2, nullptr, adj2, K2, K2, K1, K2, K2, K2,
         (long)K1*K2, (long)K1*K2, (long)K2*K2, BGR, 1, 0);              // adj2 = s^T A s
    pool_prep<<<(BGR*K2+TB-1)/TB, TB, 0, stream>>>(adj2, deg, K2);
    pool_norm<<<(int)(((long)BGR*K2*K2+TB-1)/TB), TB, 0, stream>>>(adj2, deg, K2);

    // ---- DenseGraphConv 3 + LayerNorm + ReLU ----
    gemm(stream, adj2, x2, nullptr, tmpC, K2, HID, K2, K2, HID, HID,
         (long)K2*K2, (long)K2*HID, (long)K2*HID, BGR, 0, 0);
    gemm(stream, x2, Wroot3, nullptr, tmpD, K2, HID, HID, HID, HID, HID,
         (long)K2*HID, 0, (long)K2*HID, BGR, 0, 0);
    gemm(stream, tmpC, Wrel3, brel3, tmpD, K2, HID, HID, HID, HID, HID,
         (long)K2*HID, 0, (long)K2*HID, BGR, 0, 1);
    ln_relu<<<(BGR*K2+TB-1)/TB, TB, 0, stream>>>(tmpD, ln3g, ln3b, BGR*K2);

    // ---- lin1 (+ReLU) and readout ----
    gemm(stream, tmpD, Wl1, bl1, l1o, BGR*K2, HID, HID, HID, HID, HID, 0,0,0, 1, 0, 2);
    readout<<<1, BGR, 0, stream>>>(l1o, Wro, bro, out);
}
